// RecurrentGATTrackerV3_63127429316769
// MI455X (gfx1250) — compile-verified, matrix operand order checked by reading, not run on hardware
//
#include <hip/hip_runtime.h>
#include <hip/hip_bf16.h>

// ---------------------------------------------------------------------------
// RecurrentGATTrackerV3 for MI455X (gfx1250, wave32, WMMA)
// Dense transforms -> v_wmma_f32_16x16x32_f16 with pre-packed B fragments and
// compile-time-K unrolled loops; graph scatter/gather -> L2-resident atomic
// kernels. N=50000=3125*16, E=800000, HID=64, K padded to 32 where needed.
// ---------------------------------------------------------------------------

#define NNODES 50000
#define NEDGES 800000
#define HID 64
#define NHEADS 4
#define HC 16
#define EMB 8
#define EDIM 6
#define SDIM 6
#define FEAT 7

typedef __attribute__((ext_vector_type(16))) _Float16 v16h;
typedef __attribute__((ext_vector_type(8)))  _Float16 v8h;
typedef __attribute__((ext_vector_type(8)))  float    v8f;

// ----------------------------- helpers -------------------------------------

__global__ void k_fill_f32(float* p, float v, size_t n) {
    size_t i = (size_t)blockIdx.x * blockDim.x + threadIdx.x;
    if (i < n) p[i] = v;
}

// f32 -> f16 elementwise
__global__ void k_convert(const float* __restrict__ src, _Float16* __restrict__ dst, size_t n) {
    size_t i = (size_t)blockIdx.x * blockDim.x + threadIdx.x;
    if (i < n) dst[i] = (_Float16)src[i];
}

// Pack weight B[srcRows x Nout] (f32) into WMMA B-fragment order (f16), zero-
// padding K rows up to K. Layout: dst[((nt*Kc + chunk)*32 + lane)*16 + j],
// where lane holds column n = nt*16 + (lane&15), and pair j=2i,2i+1 holds
// K index kb = (i<4 ? 2i : 16+2(i-4)) + 8*(lane>>4) + chunk*32.
__global__ void k_pack_b(const float* __restrict__ src, _Float16* __restrict__ dst,
                         int srcRows, int K, int Nout) {
    int idx = blockIdx.x * blockDim.x + threadIdx.x;
    int total = K * Nout;
    if (idx >= total) return;
    int Kc = K >> 5;
    int j    = idx & 15;
    int lane = (idx >> 4) & 31;
    int rest = idx >> 9;
    int chunk = rest % Kc;
    int nt    = rest / Kc;
    int n    = nt * 16 + (lane & 15);
    int half = lane >> 4;
    int i    = j >> 1, r = j & 1;
    int kb   = ((i < 4) ? (2 * i) : (16 + 2 * (i - 4))) + 8 * half + r;
    int k    = chunk * 32 + kb;
    dst[idx] = (k < srcRows) ? (_Float16)src[(size_t)k * Nout + n] : (_Float16)0.0f;
}

// Build padded-concat encoder input: [x(7) | type_emb(8) | sensor_emb(8) | 0(9)] as f16
__global__ void k_build_x0(const float* __restrict__ x, const int* __restrict__ ntype,
                           const int* __restrict__ sid, const float* __restrict__ temb,
                           const float* __restrict__ semb, _Float16* __restrict__ x0h) {
    int i = blockIdx.x * blockDim.x + threadIdx.x;
    if (i >= NNODES * 32) return;
    int n = i >> 5, c = i & 31;
    float v = 0.0f;
    if (c < FEAT)                v = x[(size_t)n * FEAT + c];
    else if (c < FEAT + EMB)     v = temb[(size_t)ntype[n] * EMB + (c - FEAT)];
    else if (c < FEAT + 2 * EMB) v = semb[(size_t)sid[n] * EMB + (c - FEAT - EMB)];
    x0h[i] = (_Float16)v;
}

// ----------------------------- WMMA GEMM -----------------------------------
// C[M x Nout] = A[M x K] * Bpack (+bias) (+relu). A f16 row-major, Bpack in
// fragment order (see k_pack_b). One wave computes one 16x16 output tile; the
// compile-time K unrolls into K/32 back-to-back v_wmma_f32_16x16x32_f16 ops.
// A frag per lane = two contiguous 16B chunks: Arow+8*half and Arow+16+8*half.
// C frag: lane holds col n = nt*16+(lane&15); acc[j] = C[mt*16 + j + 8*half][n].
template <int K, bool HASF, bool HASH, bool RELU>
__global__ void k_gemm_wmma(const _Float16* __restrict__ A, const _Float16* __restrict__ Bp,
                            const float* __restrict__ bias, float* __restrict__ Cf,
                            _Float16* __restrict__ Ch, int M, int Nout) {
    constexpr int KC = K / 32;
    int mt = blockIdx.x, nt = blockIdx.y;
    int lane = threadIdx.x & 31;
    int idx = lane & 15, half = lane >> 4;
    int m = mt * 16 + idx;
    if (m >= M) m = M - 1;
    int n = nt * 16 + idx;
    const _Float16* Arow = A + (size_t)m * K;
    const v16h* Bfrag = (const v16h*)Bp + ((size_t)nt * KC * 32 + lane);
    v8f acc = {};
#pragma unroll
    for (int c = 0; c < KC; ++c) {
        v8h alo = *(const v8h*)(Arow + c * 32 + 8 * half);
        v8h ahi = *(const v8h*)(Arow + c * 32 + 16 + 8 * half);
        v16h a = __builtin_shufflevector(alo, ahi, 0, 1, 2, 3, 4, 5, 6, 7,
                                         8, 9, 10, 11, 12, 13, 14, 15);
        v16h b = Bfrag[c * 32];
        acc = __builtin_amdgcn_wmma_f32_16x16x32_f16(false, a, false, b,
                                                     (short)0, acc, false, false);
    }
    float bv = bias[n];
#pragma unroll
    for (int j = 0; j < 8; ++j) {
        int row = mt * 16 + j + 8 * half;
        float v = acc[j] + bv;
        if (RELU) v = fmaxf(v, 0.0f);
        if (HASF) Cf[(size_t)row * Nout + n] = v;
        if (HASH) Ch[(size_t)row * Nout + n] = (_Float16)v;
    }
}

// ----------------------------- GAT edge kernels -----------------------------

// Per-edge attention logits: m = leaky(xl[src]+xr[dst]+edge_attr@we); logit[h]=sum m*att
__global__ void k_edge_logits(const int* __restrict__ src, const int* __restrict__ dst,
                              const float* __restrict__ eattr, const float* __restrict__ we,
                              const float* __restrict__ att, const float* __restrict__ xl,
                              const float* __restrict__ xr, float* __restrict__ logits) {
    __shared__ float sWe[EDIM * HID];
    __shared__ float sAtt[HID];
    for (int i = threadIdx.x; i < EDIM * HID; i += blockDim.x) sWe[i] = we[i];
    for (int i = threadIdx.x; i < HID; i += blockDim.x) sAtt[i] = att[i];
    __syncthreads();
    int e = blockIdx.x * blockDim.x + threadIdx.x;
    if (e >= NEDGES) return;
    int s = src[e], d = dst[e];
    float ea[EDIM];
#pragma unroll
    for (int j = 0; j < EDIM; ++j) ea[j] = eattr[(size_t)e * EDIM + j];
    const float* pl = xl + (size_t)s * HID;
    const float* pr = xr + (size_t)d * HID;
    float lg[NHEADS] = {0.f, 0.f, 0.f, 0.f};
#pragma unroll
    for (int c = 0; c < HID; ++c) {
        float xe = 0.0f;
#pragma unroll
        for (int j = 0; j < EDIM; ++j) xe += ea[j] * sWe[j * HID + c];
        float v = pl[c] + pr[c] + xe;
        v = (v > 0.0f) ? v : 0.2f * v;        // leaky_relu 0.2
        lg[c >> 4] += v * sAtt[c];            // att is [NH,HC] row-major == flat c
    }
#pragma unroll
    for (int h = 0; h < NHEADS; ++h) logits[(size_t)e * NHEADS + h] = lg[h];
}

__device__ inline void atomicMaxF(float* addr, float val) {
    if (val >= 0.0f) atomicMax((int*)addr, __float_as_int(val));
    else             atomicMin((unsigned int*)addr, (unsigned int)__float_as_int(val));
}

// segment max of logits over destination nodes
__global__ void k_segmax(const int* __restrict__ dst, const float* __restrict__ logits,
                         float* __restrict__ mx) {
    int i = blockIdx.x * blockDim.x + threadIdx.x;
    if (i >= NEDGES * NHEADS) return;
    int e = i >> 2, h = i & 3;
    atomicMaxF(&mx[(size_t)dst[e] * NHEADS + h], logits[i]);
}

// ex = exp(logit - mx[dst]) (overwrite logits); den[dst] += ex
__global__ void k_expsum(const int* __restrict__ dst, float* __restrict__ logits,
                         const float* __restrict__ mx, float* __restrict__ den) {
    int i = blockIdx.x * blockDim.x + threadIdx.x;
    if (i >= NEDGES * NHEADS) return;
    int e = i >> 2, h = i & 3;
    int d = dst[e];
    float ex = __expf(logits[i] - mx[(size_t)d * NHEADS + h]);
    logits[i] = ex;
    atomicAdd(&den[(size_t)d * NHEADS + h], ex);
}

// alpha = ex/(den+eps) -> d_out ; agg[dst] += xl[src] * alpha  (scatter aggregate)
__global__ void k_alpha_agg(const int* __restrict__ src, const int* __restrict__ dst,
                            const float* __restrict__ ex, const float* __restrict__ den,
                            const float* __restrict__ xl, float* __restrict__ alpha_out,
                            float* __restrict__ agg) {
    int e = blockIdx.x * blockDim.x + threadIdx.x;
    if (e >= NEDGES) return;
    int s = src[e], d = dst[e];
    float al[NHEADS];
#pragma unroll
    for (int h = 0; h < NHEADS; ++h) {
        al[h] = ex[(size_t)e * NHEADS + h] / (den[(size_t)d * NHEADS + h] + 1e-16f);
        alpha_out[(size_t)e * NHEADS + h] = al[h];
    }
    const float* pl = xl + (size_t)s * HID;
    float* pa = agg + (size_t)d * HID;
#pragma unroll
    for (int c = 0; c < HID; ++c) atomicAdd(&pa[c], pl[c] * al[c >> 4]);
}

// node finish: h = agg + bias (optional relu), write f16
__global__ void k_node_finish(const float* __restrict__ agg, const float* __restrict__ bias,
                              _Float16* __restrict__ outh, int relu) {
    int i = blockIdx.x * blockDim.x + threadIdx.x;
    if (i >= NNODES * HID) return;
    int c = i & (HID - 1);
    float v = agg[i] + bias[c];
    if (relu) v = fmaxf(v, 0.0f);
    outh[i] = (_Float16)v;
}

// ----------------------------- GRU + decoder --------------------------------

__global__ void k_gru(const float* __restrict__ gi, const float* __restrict__ gh,
                      const float* __restrict__ hprev, float* __restrict__ hnew,
                      _Float16* __restrict__ hnew16) {
    int i = blockIdx.x * blockDim.x + threadIdx.x;
    if (i >= NNODES * HID) return;
    int n = i >> 6, c = i & 63;
    const float* gin = gi + (size_t)n * 3 * HID;
    const float* ghn = gh + (size_t)n * 3 * HID;
    float r = 1.0f / (1.0f + __expf(-(gin[c] + ghn[c])));
    float z = 1.0f / (1.0f + __expf(-(gin[HID + c] + ghn[HID + c])));
    float nn = tanhf(gin[2 * HID + c] + r * ghn[2 * HID + c]);
    float hp = hprev[i];
    float h = (1.0f - z) * nn + z * hp;
    hnew[i] = h;
    hnew16[i] = (_Float16)h;
}

// decoder layer 2: [N,64] @ [64,7] + b  (small Nout -> scalar, w in LDS)
__global__ void k_dec2(const float* __restrict__ t, const float* __restrict__ w2,
                       const float* __restrict__ b2, float* __restrict__ out) {
    __shared__ float sW[HID * (SDIM + 1)];
    __shared__ float sB[SDIM + 1];
    for (int i = threadIdx.x; i < HID * (SDIM + 1); i += blockDim.x) sW[i] = w2[i];
    for (int i = threadIdx.x; i < SDIM + 1; i += blockDim.x) sB[i] = b2[i];
    __syncthreads();
    int n = blockIdx.x * blockDim.x + threadIdx.x;
    if (n >= NNODES) return;
    const float* tn = t + (size_t)n * HID;
    float acc[SDIM + 1];
#pragma unroll
    for (int j = 0; j < SDIM + 1; ++j) acc[j] = sB[j];
#pragma unroll
    for (int c = 0; c < HID; ++c) {
        float tv = tn[c];
#pragma unroll
        for (int j = 0; j < SDIM + 1; ++j) acc[j] += tv * sW[c * (SDIM + 1) + j];
    }
#pragma unroll
    for (int j = 0; j < SDIM + 1; ++j) out[(size_t)n * (SDIM + 1) + j] = acc[j];
}

// ----------------------------- launch --------------------------------------

static inline size_t wsAlloc(size_t& off, size_t bytes) {
    size_t p = off;
    off += (bytes + 255) & ~(size_t)255;
    return p;
}

extern "C" void kernel_launch(void* const* d_in, const int* in_sizes, int n_in,
                              void* d_out, int out_size, void* d_ws, size_t ws_size,
                              hipStream_t stream) {
    (void)in_sizes; (void)n_in; (void)out_size; (void)ws_size;
    const float* x          = (const float*)d_in[0];
    const int*   node_type  = (const int*)d_in[1];
    const int*   sensor_id  = (const int*)d_in[2];
    const int*   edge_index = (const int*)d_in[3];
    const float* edge_attr  = (const float*)d_in[4];
    const float* hidden     = (const float*)d_in[5];
    const float* type_emb   = (const float*)d_in[6];
    const float* sensor_emb = (const float*)d_in[7];
    const float* enc_w1 = (const float*)d_in[8];  const float* enc_b1 = (const float*)d_in[9];
    const float* enc_w2 = (const float*)d_in[10]; const float* enc_b2 = (const float*)d_in[11];
    const float* g1_wl = (const float*)d_in[12]; const float* g1_bl = (const float*)d_in[13];
    const float* g1_wr = (const float*)d_in[14]; const float* g1_br = (const float*)d_in[15];
    const float* g1_we = (const float*)d_in[16]; const float* g1_att = (const float*)d_in[17];
    const float* g1_bias = (const float*)d_in[18];
    const float* g2_wl = (const float*)d_in[19]; const float* g2_bl = (const float*)d_in[20];
    const float* g2_wr = (const float*)d_in[21]; const float* g2_br = (const float*)d_in[22];
    const float* g2_we = (const float*)d_in[23]; const float* g2_att = (const float*)d_in[24];
    const float* g2_bias = (const float*)d_in[25];
    const float* gru_wih = (const float*)d_in[26]; const float* gru_bih = (const float*)d_in[27];
    const float* gru_whh = (const float*)d_in[28]; const float* gru_bhh = (const float*)d_in[29];
    const float* dec_w1 = (const float*)d_in[30]; const float* dec_b1 = (const float*)d_in[31];
    const float* dec_w2 = (const float*)d_in[32]; const float* dec_b2 = (const float*)d_in[33];

    const int* src = edge_index;            // edge_index[0]
    const int* dst = edge_index + NEDGES;   // edge_index[1]

    float* out_pred  = (float*)d_out;                                         // [N,7]
    float* out_hnew  = (float*)d_out + (size_t)NNODES * (SDIM + 1);           // [N,64]
    float* out_a1    = out_hnew + (size_t)NNODES * HID;                       // [E,4]
    float* out_a2    = out_a1 + (size_t)NEDGES * NHEADS;                      // [E,4]

    // ---- workspace layout ----
    char* ws = (char*)d_ws;
    size_t off = 0;
    _Float16* x0h    = (_Float16*)(ws + wsAlloc(off, (size_t)NNODES * 32 * 2));
    _Float16* h16a   = (_Float16*)(ws + wsAlloc(off, (size_t)NNODES * HID * 2));
    _Float16* h16b   = (_Float16*)(ws + wsAlloc(off, (size_t)NNODES * HID * 2));
    _Float16* hid16  = (_Float16*)(ws + wsAlloc(off, (size_t)NNODES * HID * 2));
    _Float16* nh16   = (_Float16*)(ws + wsAlloc(off, (size_t)NNODES * HID * 2));
    float* xl        = (float*)(ws + wsAlloc(off, (size_t)NNODES * HID * 4));
    float* xr        = (float*)(ws + wsAlloc(off, (size_t)NNODES * HID * 4));
    float* agg       = (float*)(ws + wsAlloc(off, (size_t)NNODES * HID * 4));
    float* gi        = (float*)(ws + wsAlloc(off, (size_t)NNODES * 3 * HID * 4));
    float* gh       = (float*)(ws + wsAlloc(off, (size_t)NNODES * 3 * HID * 4));
    float* logits    = (float*)(ws + wsAlloc(off, (size_t)NEDGES * NHEADS * 4));
    float* mx        = (float*)(ws + wsAlloc(off, (size_t)NNODES * NHEADS * 4));
    float* den       = (float*)(ws + wsAlloc(off, (size_t)NNODES * NHEADS * 4));
    float* dec_t     = (float*)(ws + wsAlloc(off, (size_t)NNODES * HID * 4));
    _Float16* w_e1p  = (_Float16*)(ws + wsAlloc(off, 32 * 64 * 2));
    _Float16* w_e2p  = (_Float16*)(ws + wsAlloc(off, 64 * 64 * 2));
    _Float16* w1lp   = (_Float16*)(ws + wsAlloc(off, 64 * 64 * 2));
    _Float16* w1rp   = (_Float16*)(ws + wsAlloc(off, 64 * 64 * 2));
    _Float16* w2lp   = (_Float16*)(ws + wsAlloc(off, 64 * 64 * 2));
    _Float16* w2rp   = (_Float16*)(ws + wsAlloc(off, 64 * 64 * 2));
    _Float16* wihp   = (_Float16*)(ws + wsAlloc(off, 64 * 192 * 2));
    _Float16* whhp   = (_Float16*)(ws + wsAlloc(off, 64 * 192 * 2));
    _Float16* wd1p   = (_Float16*)(ws + wsAlloc(off, 64 * 64 * 2));

    const int TB = 256;
    const int MT = NNODES / 16;            // 3125 row tiles
    dim3 b32(32);

    // ---- weight packing (f32 -> f16 WMMA B-fragment order, K-padded) ----
    k_pack_b<<<(32 * 64 + TB - 1) / TB, TB, 0, stream>>>(enc_w1, w_e1p, 23, 32, 64);
    k_pack_b<<<(64 * 64 + TB - 1) / TB, TB, 0, stream>>>(enc_w2, w_e2p, 64, 64, 64);
    k_pack_b<<<(64 * 64 + TB - 1) / TB, TB, 0, stream>>>(g1_wl, w1lp, 64, 64, 64);
    k_pack_b<<<(64 * 64 + TB - 1) / TB, TB, 0, stream>>>(g1_wr, w1rp, 64, 64, 64);
    k_pack_b<<<(64 * 64 + TB - 1) / TB, TB, 0, stream>>>(g2_wl, w2lp, 64, 64, 64);
    k_pack_b<<<(64 * 64 + TB - 1) / TB, TB, 0, stream>>>(g2_wr, w2rp, 64, 64, 64);
    k_pack_b<<<(64 * 192 + TB - 1) / TB, TB, 0, stream>>>(gru_wih, wihp, 64, 64, 192);
    k_pack_b<<<(64 * 192 + TB - 1) / TB, TB, 0, stream>>>(gru_whh, whhp, 64, 64, 192);
    k_pack_b<<<(64 * 64 + TB - 1) / TB, TB, 0, stream>>>(dec_w1, wd1p, 64, 64, 64);
    k_convert<<<((size_t)NNODES * HID + TB - 1) / TB, TB, 0, stream>>>(hidden, hid16, (size_t)NNODES * HID);

    // ---- encoder ----
    k_build_x0<<<(NNODES * 32 + TB - 1) / TB, TB, 0, stream>>>(x, node_type, sensor_id,
                                                               type_emb, sensor_emb, x0h);
    k_gemm_wmma<32, false, true, true><<<dim3(MT, 4), b32, 0, stream>>>(
        x0h, w_e1p, enc_b1, nullptr, h16a, NNODES, 64);
    k_gemm_wmma<64, false, true, false><<<dim3(MT, 4), b32, 0, stream>>>(
        h16a, w_e2p, enc_b2, nullptr, h16b, NNODES, 64);

    // ---- GAT layer 1 ----
    k_gemm_wmma<64, true, false, false><<<dim3(MT, 4), b32, 0, stream>>>(
        h16b, w1lp, g1_bl, xl, nullptr, NNODES, 64);
    k_gemm_wmma<64, true, false, false><<<dim3(MT, 4), b32, 0, stream>>>(
        h16b, w1rp, g1_br, xr, nullptr, NNODES, 64);
    k_fill_f32<<<((size_t)NNODES * NHEADS + TB - 1) / TB, TB, 0, stream>>>(mx, -INFINITY, (size_t)NNODES * NHEADS);
    k_fill_f32<<<((size_t)NNODES * NHEADS + TB - 1) / TB, TB, 0, stream>>>(den, 0.0f, (size_t)NNODES * NHEADS);
    k_fill_f32<<<((size_t)NNODES * HID + TB - 1) / TB, TB, 0, stream>>>(agg, 0.0f, (size_t)NNODES * HID);
    k_edge_logits<<<(NEDGES + TB - 1) / TB, TB, 0, stream>>>(src, dst, edge_attr, g1_we, g1_att, xl, xr, logits);
    k_segmax<<<((size_t)NEDGES * NHEADS + TB - 1) / TB, TB, 0, stream>>>(dst, logits, mx);
    k_expsum<<<((size_t)NEDGES * NHEADS + TB - 1) / TB, TB, 0, stream>>>(dst, logits, mx, den);
    k_alpha_agg<<<(NEDGES + TB - 1) / TB, TB, 0, stream>>>(src, dst, logits, den, xl, out_a1, agg);
    k_node_finish<<<((size_t)NNODES * HID + TB - 1) / TB, TB, 0, stream>>>(agg, g1_bias, h16a, 1);

    // ---- GAT layer 2 ----
    k_gemm_wmma<64, true, false, false><<<dim3(MT, 4), b32, 0, stream>>>(
        h16a, w2lp, g2_bl, xl, nullptr, NNODES, 64);
    k_gemm_wmma<64, true, false, false><<<dim3(MT, 4), b32, 0, stream>>>(
        h16a, w2rp, g2_br, xr, nullptr, NNODES, 64);
    k_fill_f32<<<((size_t)NNODES * NHEADS + TB - 1) / TB, TB, 0, stream>>>(mx, -INFINITY, (size_t)NNODES * NHEADS);
    k_fill_f32<<<((size_t)NNODES * NHEADS + TB - 1) / TB, TB, 0, stream>>>(den, 0.0f, (size_t)NNODES * NHEADS);
    k_fill_f32<<<((size_t)NNODES * HID + TB - 1) / TB, TB, 0, stream>>>(agg, 0.0f, (size_t)NNODES * HID);
    k_edge_logits<<<(NEDGES + TB - 1) / TB, TB, 0, stream>>>(src, dst, edge_attr, g2_we, g2_att, xl, xr, logits);
    k_segmax<<<((size_t)NEDGES * NHEADS + TB - 1) / TB, TB, 0, stream>>>(dst, logits, mx);
    k_expsum<<<((size_t)NEDGES * NHEADS + TB - 1) / TB, TB, 0, stream>>>(dst, logits, mx, den);
    k_alpha_agg<<<(NEDGES + TB - 1) / TB, TB, 0, stream>>>(src, dst, logits, den, xl, out_a2, agg);
    k_node_finish<<<((size_t)NNODES * HID + TB - 1) / TB, TB, 0, stream>>>(agg, g2_bias, h16b, 0);

    // ---- GRU ----
    k_gemm_wmma<64, true, false, false><<<dim3(MT, 12), b32, 0, stream>>>(
        h16b, wihp, gru_bih, gi, nullptr, NNODES, 192);
    k_gemm_wmma<64, true, false, false><<<dim3(MT, 12), b32, 0, stream>>>(
        hid16, whhp, gru_bhh, gh, nullptr, NNODES, 192);
    k_gru<<<((size_t)NNODES * HID + TB - 1) / TB, TB, 0, stream>>>(gi, gh, hidden, out_hnew, nh16);

    // ---- decoder ----
    k_gemm_wmma<64, true, false, true><<<dim3(MT, 4), b32, 0, stream>>>(
        nh16, wd1p, dec_b1, dec_t, nullptr, NNODES, 64);
    k_dec2<<<(NNODES + TB - 1) / TB, TB, 0, stream>>>(dec_t, dec_w2, dec_b2, out_pred);
}
